// MultiPrecActivConv1d_13812614824157
// MI455X (gfx1250) — matrix operand order repaired and checked by reading, not compiled, not run on hardware
//
#include <hip/hip_runtime.h>
#include <cmath>

// ---------------------------------------------------------------------------
// MixPrec PACT conv1d forward for gfx1250 (MI455X), wave32 + WMMA f16->f32.
// B=16, CIN=64, COUT=64, L=16384, K=3, pad=1.
// ---------------------------------------------------------------------------

typedef _Float16 h16;
typedef __attribute__((ext_vector_type(16))) _Float16 v16h;
typedef __attribute__((ext_vector_type(8)))  _Float16 v8h;
typedef __attribute__((ext_vector_type(8)))  float    v8f;

#define B_      16
#define CIN_    64
#define COUT_   64
#define L_      16384
#define KW_     3
#define TILE_P  256
#define NP      258              // TILE_P + 2 halo positions

// workspace layout:
//   [0, 24576)       : mixed f16 weights, [k][cout][cin]
//   [24576, 24832)   : quantized bias, f32 x64

// ---------------------------------------------------------------------------
// Prep: per-channel min/max fake-quant of weights at 2/4/8 bits, softmax mix,
// bias quant, and the w_complexity scalar.
// ---------------------------------------------------------------------------
__global__ void mixprec_prep_kernel(const float* __restrict__ w,
                                    const float* __restrict__ bias,
                                    const float* __restrict__ alpha_w,
                                    const int*   __restrict__ temp_p,
                                    h16*   __restrict__ ws_w,
                                    float* __restrict__ ws_qb,
                                    float* __restrict__ wcomp_out) {
  const int c = threadIdx.x;                      // output channel 0..63
  const float temp = (float)(*temp_p);

  // per-channel range over CIN*K = 192 weights
  const float* wr = w + c * (CIN_ * KW_);
  float mn = wr[0], mx = wr[0];
  for (int i = 1; i < CIN_ * KW_; ++i) {
    float v = wr[i];
    mn = fminf(mn, v);
    mx = fmaxf(mx, v);
  }
  float rng = mx - mn;
  if (rng == 0.f) rng = 1.f;

  // per-channel softmax over the bit axis
  float a0 = alpha_w[0 * COUT_ + c] / temp;
  float a1 = alpha_w[1 * COUT_ + c] / temp;
  float a2 = alpha_w[2 * COUT_ + c] / temp;
  float m  = fmaxf(a0, fmaxf(a1, a2));
  float e0 = expf(a0 - m), e1 = expf(a1 - m), e2 = expf(a2 - m);
  float inv = 1.f / (e0 + e1 + e2);
  float sw0 = e0 * inv, sw1 = e1 * inv, sw2 = e2 * inv;

  float s0 = rng * (1.f / 3.f);                   // 2-bit
  float s1 = rng * (1.f / 15.f);                  // 4-bit
  float s2 = rng * (1.f / 255.f);                 // 8-bit

  for (int i = 0; i < CIN_ * KW_; ++i) {
    float v = wr[i];
    float q = sw0 * (roundf(v / s0) * s0)
            + sw1 * (roundf(v / s1) * s1)
            + sw2 * (roundf(v / s2) * s2);
    int k  = i % KW_;
    int ci = i / KW_;
    ws_w[((size_t)k * COUT_ + c) * CIN_ + ci] = (h16)q;   // [k][cout][cin]
  }

  // bias fake-quant (bit = 32)
  float bmn = bias[0], bmx = bias[0];
  for (int i = 1; i < COUT_; ++i) {
    float v = bias[i];
    bmn = fminf(bmn, v);
    bmx = fmaxf(bmx, v);
  }
  float bs = (bmx - bmn) * (1.0f / 4294967295.0f);
  ws_qb[c] = (bs != 0.f) ? roundf(bias[c] / bs) * bs : bias[c];

  // effective-bitwidth complexity
  __shared__ float red[COUT_];
  red[c] = sw0 * 2.f + sw1 * 4.f + sw2 * 8.f;
  __syncthreads();
  if (c == 0) {
    float s = 0.f;
    for (int i = 0; i < COUT_; ++i) s += red[i];
    *wcomp_out = (float)(CIN_ * COUT_ * (KW_ * KW_) * 1e-6) * (s / (float)COUT_);
  }
}

// ---------------------------------------------------------------------------
// Main: PACT-mix x on the fly into f16 LDS tiles, then implicit-GEMM conv via
// v_wmma_f32_16x16x32_f16 (3 k-taps x 2 cin-halves = 6 k-steps of 32).
// Block = 256 threads (8 wave32), tile = 64 cout x 256 positions.
// ---------------------------------------------------------------------------
__global__ void __launch_bounds__(256)
mixprec_conv_wmma_kernel(const float* __restrict__ x,
                         const float* __restrict__ clip_vals,
                         const float* __restrict__ alpha_a,
                         const h16*   __restrict__ ws_w,
                         const float* __restrict__ ws_qb,
                         float* __restrict__ out) {
  __shared__ __align__(32) h16 lds_act[NP * CIN_];        // [pos][cin], ~33 KB
  __shared__ __align__(16) h16 lds_w[KW_ * COUT_ * CIN_]; // [k][cout][cin], ~24.5 KB

  const int tid  = threadIdx.x;
  const int bidx = blockIdx.x;
  const int b    = bidx >> 6;                  // 64 tiles per batch row
  const int l0   = (bidx & 63) * TILE_P;

  // activation softmax mix weights + PACT scales (uniform -> scalar regs)
  float a0 = alpha_a[0], a1 = alpha_a[1], a2 = alpha_a[2];
  float m  = fmaxf(a0, fmaxf(a1, a2));
  float e0 = expf(a0 - m), e1 = expf(a1 - m), e2 = expf(a2 - m);
  float si = 1.f / (e0 + e1 + e2);
  float sa0 = e0 * si, sa1 = e1 * si, sa2 = e2 * si;
  float c0 = clip_vals[0], c1 = clip_vals[1], c2 = clip_vals[2];
  float sc0 = 3.f / c0,   sc1 = 15.f / c1,   sc2 = 255.f / c2;
  float is0 = c0 * (1.f / 3.f), is1 = c1 * (1.f / 15.f), is2 = c2 * (1.f / 255.f);

  // stage mixed weights to LDS with 128-bit copies
  {
    const uint4* src = reinterpret_cast<const uint4*>(ws_w);
    uint4*       dst = reinterpret_cast<uint4*>(lds_w);
    for (int i = tid; i < (KW_ * COUT_ * CIN_ * 2) / 16; i += 256) dst[i] = src[i];
  }

  // stage PACT-quantized activations: x[b][ci][l0-1 .. l0+256] -> lds[p][ci]
  for (int idx = tid; idx < CIN_ * NP; idx += 256) {
    int ci = idx / NP;
    int p  = idx - ci * NP;
    int l  = l0 - 1 + p;
    float xv = 0.f;
    if ((unsigned)l < (unsigned)L_)
      xv = x[((size_t)b * CIN_ + ci) * L_ + l];
    float y0 = fminf(fmaxf(xv, 0.f), c0);
    float y1 = fminf(fmaxf(xv, 0.f), c1);
    float y2 = fminf(fmaxf(xv, 0.f), c2);
    float act = sa0 * (floorf(y0 * sc0) * is0)
              + sa1 * (floorf(y1 * sc1) * is1)
              + sa2 * (floorf(y2 * sc2) * is2);
    lds_act[p * CIN_ + ci] = (h16)act;
  }
  __syncthreads();

  const int wave = tid >> 5;
  const int lane = tid & 31;
  const int lh   = lane >> 4;        // lane half (ISA A/B/C layouts split at 16)
  const int ln   = lane & 15;
  const int ct   = wave & 3;         // cout tile (16 rows each)
  const int pg   = wave >> 2;        // position group (128 cols each)

  v8f acc[8];
#pragma unroll
  for (int j = 0; j < 8; ++j)
    acc[j] = (v8f){0.f, 0.f, 0.f, 0.f, 0.f, 0.f, 0.f, 0.f};

#pragma unroll
  for (int k = 0; k < KW_; ++k) {
#pragma unroll
    for (int hh = 0; hh < 2; ++hh) {            // cin halves: K-step of 32
      // A fragment (weights 16x32): lane row M = ln, 16 contiguous K per lane
      int arow = ct * 16 + ln;
      int ci0  = hh * 32 + lh * 16;
      const h16* abase = &lds_w[((size_t)k * COUT_ + arow) * CIN_ + ci0];
      v8h alo = *reinterpret_cast<const v8h*>(abase);
      v8h ahi = *reinterpret_cast<const v8h*>(abase + 8);
      v16h afrag;
#pragma unroll
      for (int e = 0; e < 8; ++e) { afrag[e] = alo[e]; afrag[8 + e] = ahi[e]; }

      int bci0 = hh * 32 + lh * 16;             // B: 16 contiguous K per lane
#pragma unroll
      for (int j = 0; j < 8; ++j) {
        int p = pg * 128 + j * 16 + ln + k;     // halo'd LDS row (l = l0-1+p)
        v16h bfrag = *reinterpret_cast<const v16h*>(&lds_act[p * CIN_ + bci0]);
        acc[j] = __builtin_amdgcn_wmma_f32_16x16x32_f16(
            false, afrag, false, bfrag, (short)0, acc[j], false, false);
      }
    }
  }

  // epilogue: C layout (lane l, vgpr r) -> M = r + lh*8, N = ln; add bias
  float qbv[8];
#pragma unroll
  for (int r = 0; r < 8; ++r) qbv[r] = ws_qb[ct * 16 + lh * 8 + r];
#pragma unroll
  for (int j = 0; j < 8; ++j) {
    int col = l0 + pg * 128 + j * 16 + ln;
#pragma unroll
    for (int r = 0; r < 8; ++r) {
      int crow = ct * 16 + lh * 8 + r;
      out[((size_t)b * COUT_ + crow) * L_ + col] = acc[j][r] + qbv[r];
    }
  }
}

// ---------------------------------------------------------------------------
extern "C" void kernel_launch(void* const* d_in, const int* in_sizes, int n_in,
                              void* d_out, int out_size, void* d_ws, size_t ws_size,
                              hipStream_t stream) {
  (void)in_sizes; (void)n_in; (void)ws_size;
  const float* x      = (const float*)d_in[0];
  const float* w      = (const float*)d_in[1];
  const float* bias   = (const float*)d_in[2];
  const float* clipv  = (const float*)d_in[3];
  const float* aact   = (const float*)d_in[4];
  const float* awght  = (const float*)d_in[5];
  const int*   temp   = (const int*)d_in[6];

  float* out   = (float*)d_out;
  h16*   ws_w  = (h16*)d_ws;
  float* ws_qb = (float*)((char*)d_ws + (size_t)KW_ * COUT_ * CIN_ * sizeof(h16));
  float* wcomp = out + (size_t)out_size - 1;

  hipLaunchKernelGGL(mixprec_prep_kernel, dim3(1), dim3(COUT_), 0, stream,
                     w, bias, awght, temp, ws_w, ws_qb, wcomp);
  hipLaunchKernelGGL(mixprec_conv_wmma_kernel, dim3(B_ * (L_ / TILE_P)), dim3(256),
                     0, stream, x, clipv, aact, ws_w, ws_qb, out);
}
